// ANPToolEncoder_4638564680178
// MI455X (gfx1250) — matrix operand order repaired and checked by reading, not compiled, not run on hardware
//
#include <hip/hip_runtime.h>
#include <hip/hip_bf16.h>
#include <math.h>

// Problem constants (match reference)
#define BB   256
#define MM   64
#define CC   256
#define DXX  512
#define HH   512
#define NHH  8
#define DHH  64
#define MC_  (MM * CC)     // 16384 tokens
#define KP   544           // DX+2=514 padded up to a multiple of 32
#define EPS_LN 1e-5f

typedef __bf16 v16bf __attribute__((ext_vector_type(16)));
typedef __bf16 v8bf  __attribute__((ext_vector_type(8)));
typedef float  v8f   __attribute__((ext_vector_type(8)));
typedef float  v4f   __attribute__((ext_vector_type(4)));

static __device__ __forceinline__ v8f wmma_bf16(v16bf a, v16bf b, v8f c) {
    return __builtin_amdgcn_wmma_f32_16x16x32_bf16(
        /*neg_a=*/false, a, /*neg_b=*/false, b,
        /*c_mod=*/(short)0, c, /*reuse_a=*/false, /*reuse_b=*/false);
}

// ---------------------------------------------------------------------------
// Pack kernels: zero-padded concat input [MC, 544] and padded cp_w1 [512,544]
// ---------------------------------------------------------------------------
__global__ void pack_ctxin_kernel(const float* __restrict__ x,
                                  const float* __restrict__ gt,
                                  const float* __restrict__ pred,
                                  float* __restrict__ out) {
    int i = blockIdx.x * blockDim.x + threadIdx.x;
    const int total = MC_ * KP;
    if (i >= total) return;
    int row = i / KP;
    int col = i - row * KP;
    float v;
    if (col < DXX)           v = x[(size_t)row * DXX + col];
    else if (col == DXX)     v = gt[row];
    else if (col == DXX + 1) v = pred[row];
    else                     v = 0.f;
    out[i] = v;
}

__global__ void pack_w1_kernel(const float* __restrict__ w,
                               float* __restrict__ out) {
    int i = blockIdx.x * blockDim.x + threadIdx.x;
    const int total = HH * KP;
    if (i >= total) return;
    int row = i / KP;
    int col = i - row * KP;
    out[i] = (col < DXX + 2) ? w[(size_t)row * (DXX + 2) + col] : 0.f;
}

// ---------------------------------------------------------------------------
// Batched tile transpose: in [batch][R=256][Cc=512] -> out [batch][512][256].
// Coalesced on both sides via a 32x33 LDS tile. grid=(512/32, 256/32, batch),
// block=(32,8).
// ---------------------------------------------------------------------------
__global__ __launch_bounds__(256)
void transpose_256x512_kernel(const float* __restrict__ in,
                              float* __restrict__ out) {
    __shared__ float t[32][33];
    const int m  = blockIdx.z;
    const int h0 = blockIdx.x * 32;
    const int c0 = blockIdx.y * 32;
    const float* src = in + (size_t)m * CC * HH;
    float* dst = out + (size_t)m * HH * CC;
#pragma unroll
    for (int j = 0; j < 4; ++j) {
        const int c = c0 + threadIdx.y + j * 8;
        t[threadIdx.y + j * 8][threadIdx.x] = src[(size_t)c * HH + h0 + threadIdx.x];
    }
    __syncthreads();
#pragma unroll
    for (int j = 0; j < 4; ++j) {
        const int h = h0 + threadIdx.y + j * 8;
        dst[(size_t)h * CC + c0 + threadIdx.x] = t[threadIdx.x][threadIdx.y + j * 8];
    }
}

// ---------------------------------------------------------------------------
// WMMA GEMM: OUT[i,n] = act(alpha * sum_k A[i,k]*W[n,k] + bias[n])
//   A: f32, row stride lda, batch stride aBatch (elements)
//   W: f32, row-major [N, K] with row stride ldbn (k contiguous)
//   OUT: OutT (float or __bf16), row stride ldo, batch stride oBatch
// Block: 256 threads = 8 waves; block tile 64 rows x 128 cols;
// wave tile 16 rows x 64 cols (4 WMMA accumulators per A fragment).
// grid = (N/128, M/64, batches). K must be a multiple of 32.
// ---------------------------------------------------------------------------
template <typename OutT>
__global__ __launch_bounds__(256)
void gemm_wmma_kernel(const float* __restrict__ A, int lda, int aBatch,
                      const float* __restrict__ Bm, int ldbn, int bBatch,
                      const float* __restrict__ bias,
                      float alpha, int relu, int K,
                      OutT* __restrict__ out, int ldo, int oBatch) {
    const int lane = threadIdx.x & 31;
    const int wave = threadIdx.x >> 5;
    const int lg   = lane >> 4;     // 0/1 half-wave group
    const int ln   = lane & 15;
    const int rowBase = blockIdx.y * 64 + (wave >> 1) * 16;
    const int colBase = blockIdx.x * 128 + (wave & 1) * 64;

    A  += (size_t)blockIdx.z * aBatch;
    Bm += (size_t)blockIdx.z * bBatch;

    v8f acc[4];
#pragma unroll
    for (int t = 0; t < 4; ++t)
#pragma unroll
        for (int j = 0; j < 8; ++j) acc[t][j] = 0.f;

    const float* aRow = A + (size_t)(rowBase + ln) * lda;
    const float* bRow[4];
#pragma unroll
    for (int t = 0; t < 4; ++t)
        bRow[t] = Bm + (size_t)(colBase + t * 16 + ln) * ldbn;

    for (int k0 = 0; k0 < K; k0 += 32) {
        // A fragment: lane holds row (rowBase+ln), K chunks [c, c+8) and
        // [c+16, c+24), c = k0 + lg*8  (ISA 16-bit A 16x32 layout)
        v16bf afrag;
        {
            const float* ap = aRow + k0 + lg * 8;
            __builtin_prefetch(ap + 32, 0, 0);   // global_prefetch_b8, next tile
            v4f x0 = *(const v4f*)(ap);
            v4f x1 = *(const v4f*)(ap + 4);
            v4f x2 = *(const v4f*)(ap + 16);
            v4f x3 = *(const v4f*)(ap + 20);
#pragma unroll
            for (int j = 0; j < 4; ++j) {
                afrag[j]      = (__bf16)x0[j];
                afrag[4 + j]  = (__bf16)x1[j];
                afrag[8 + j]  = (__bf16)x2[j];
                afrag[12 + j] = (__bf16)x3[j];
            }
        }
#pragma unroll
        for (int t = 0; t < 4; ++t) {
            // B fragment: lane holds column n, K chunks [c, c+8) and [c+16, c+24), c = k0 + lg*8
            const float* bp = bRow[t] + k0 + lg * 8;
            __builtin_prefetch(bp + 32, 0, 0);
            v16bf bfrag;
            v4f y0 = *(const v4f*)(bp);
            v4f y1 = *(const v4f*)(bp + 4);
            v4f y2 = *(const v4f*)(bp + 16);
            v4f y3 = *(const v4f*)(bp + 20);
#pragma unroll
            for (int j = 0; j < 4; ++j) {
                bfrag[j]      = (__bf16)y0[j];
                bfrag[4 + j]  = (__bf16)y1[j];
                bfrag[8 + j]  = (__bf16)y2[j];
                bfrag[12 + j] = (__bf16)y3[j];
            }
            acc[t] = wmma_bf16(afrag, bfrag, acc[t]);
        }
    }

    // Epilogue. D layout: reg j -> row rowBase + j + 8*lg, col = n.
#pragma unroll
    for (int t = 0; t < 4; ++t) {
        const int n = colBase + t * 16 + ln;
        const float bv = bias ? bias[n] : 0.f;
#pragma unroll
        for (int j = 0; j < 8; ++j) {
            float v = acc[t][j] * alpha + bv;
            if (relu) v = fmaxf(v, 0.f);
            const int row = rowBase + j + 8 * lg;
            const size_t idx = (size_t)row * ldo + n + (size_t)blockIdx.z * oBatch;
            out[idx] = (OutT)v;
        }
    }
}

// ---------------------------------------------------------------------------
// Fused self-attention over C=256 tokens, one block per (m, head).
// Block = 512 threads = 16 waves; wave w owns query strip [w*16, w*16+16).
// K staged in LDS as [128 keys][64 d] bf16, V transposed to [64 d][128 keys].
// Two key-halves of 128; flash-style online softmax; P transposed via per-wave
// LDS scratch to WMMA A-layout. Writes attention output (pre out-proj) as f32.
// ---------------------------------------------------------------------------
__global__ __launch_bounds__(512)
void flash_self_attn_kernel(const __bf16* __restrict__ qkv,
                            float* __restrict__ out) {
    __shared__ __bf16 kBuf[128 * 64];   // 16 KB
    __shared__ __bf16 vT[64 * 128];     // 16 KB
    __shared__ __bf16 pScr[16 * 512];   // 16 KB, per-wave 16x32 tiles

    const int m    = blockIdx.x / NHH;
    const int h    = blockIdx.x % NHH;
    const int tid  = threadIdx.x;
    const int lane = tid & 31;
    const int wave = tid >> 5;          // 0..15
    const int lg   = lane >> 4;
    const int ln   = lane & 15;
    const int q0   = wave * 16;

    const size_t rs = 3 * HH;           // 1536 bf16 per token row
    const __bf16* base = qkv + (size_t)m * CC * rs + (size_t)h * DHH;

    // Q fragments (d 0..31 and 32..63) for this wave's strip
    v16bf qa0, qa1;
    {
        const __bf16* qp = base + (size_t)(q0 + ln) * rs;
        v8bf a = *(const v8bf*)(qp + lg * 8);
        v8bf b = *(const v8bf*)(qp + lg * 8 + 16);
        v8bf c = *(const v8bf*)(qp + 32 + lg * 8);
        v8bf d = *(const v8bf*)(qp + 32 + lg * 8 + 16);
#pragma unroll
        for (int j = 0; j < 8; ++j) {
            qa0[j] = a[j]; qa0[8 + j] = b[j];
            qa1[j] = c[j]; qa1[8 + j] = d[j];
        }
    }

    float mrow[8], lrow[8];
    v8f acc[4];
#pragma unroll
    for (int r = 0; r < 8; ++r) { mrow[r] = -INFINITY; lrow[r] = 0.f; }
#pragma unroll
    for (int t = 0; t < 4; ++t)
#pragma unroll
        for (int j = 0; j < 8; ++j) acc[t][j] = 0.f;

    __bf16* pw = pScr + wave * 512;

    for (int half = 0; half < 2; ++half) {
        const int keyBase = half * 128;
        __syncthreads();   // everyone done with previous half's tiles
        {
            // Cooperative fill: 512 threads, 4 per key row, 16 d-elems each.
            const int key = tid >> 2;
            const int dc  = (tid & 3) * 16;
            const __bf16* kp = base + (size_t)(keyBase + key) * rs + HH + dc;
            v8bf k0v = *(const v8bf*)(kp);
            v8bf k1v = *(const v8bf*)(kp + 8);
            *(v8bf*)(&kBuf[key * 64 + dc])     = k0v;
            *(v8bf*)(&kBuf[key * 64 + dc + 8]) = k1v;
            const __bf16* vp = base + (size_t)(keyBase + key) * rs + 2 * HH + dc;
            v8bf v0v = *(const v8bf*)(vp);
            v8bf v1v = *(const v8bf*)(vp + 8);
#pragma unroll
            for (int j = 0; j < 8; ++j) {
                vT[(dc + j) * 128 + key]     = v0v[j];
                vT[(dc + 8 + j) * 128 + key] = v1v[j];
            }
        }
        __syncthreads();

        for (int kt = 0; kt < 128; kt += 32) {
            // ---- scores for two 16-key tiles (keys kt..kt+31, local) ----
            v8f S[2];
#pragma unroll
            for (int tt = 0; tt < 2; ++tt) {
                const int n = kt + tt * 16 + ln;       // key (B-matrix column)
                v16bf kb0, kb1;
                {
                    const __bf16* p0 = &kBuf[n * 64 + lg * 8];
                    v8bf a = *(const v8bf*)(p0);
                    v8bf b = *(const v8bf*)(p0 + 16);
                    const __bf16* p1 = &kBuf[n * 64 + 32 + lg * 8];
                    v8bf c = *(const v8bf*)(p1);
                    v8bf d = *(const v8bf*)(p1 + 16);
#pragma unroll
                    for (int j = 0; j < 8; ++j) {
                        kb0[j] = a[j]; kb0[8 + j] = b[j];
                        kb1[j] = c[j]; kb1[8 + j] = d[j];
                    }
                }
                v8f z;
#pragma unroll
                for (int j = 0; j < 8; ++j) z[j] = 0.f;
                z = wmma_bf16(qa0, kb0, z);
                z = wmma_bf16(qa1, kb1, z);
                S[tt] = z;
            }
            // ---- online softmax (rows live across 16-lane groups) ----
#pragma unroll
            for (int r = 0; r < 8; ++r) {
                float s0 = S[0][r] * 0.125f;           // DH^-0.5 = 1/8
                float s1 = S[1][r] * 0.125f;
                float t = fmaxf(s0, s1);
                t = fmaxf(t, __shfl_xor(t, 1));
                t = fmaxf(t, __shfl_xor(t, 2));
                t = fmaxf(t, __shfl_xor(t, 4));
                t = fmaxf(t, __shfl_xor(t, 8));
                float newm = fmaxf(mrow[r], t);
                float corr = __expf(mrow[r] - newm);
                mrow[r] = newm;
                float p0 = __expf(s0 - newm);
                float p1 = __expf(s1 - newm);
                float rsum = p0 + p1;
                rsum += __shfl_xor(rsum, 1);
                rsum += __shfl_xor(rsum, 2);
                rsum += __shfl_xor(rsum, 4);
                rsum += __shfl_xor(rsum, 8);
                lrow[r] = lrow[r] * corr + rsum;
#pragma unroll
                for (int t4 = 0; t4 < 4; ++t4) acc[t4][r] *= corr;
                const int prow = r + 8 * lg;
                pw[prow * 32 + ln]      = (__bf16)p0;
                pw[prow * 32 + 16 + ln] = (__bf16)p1;
            }
            asm volatile("" ::: "memory");  // order per-wave LDS store->load
            // ---- read P back in A-fragment layout (16 q x 32 keys) ----
            v16bf pa;
            {
                const __bf16* pp = pw + ln * 32 + lg * 8;
                v8bf x = *(const v8bf*)(pp);
                v8bf y = *(const v8bf*)(pp + 16);
#pragma unroll
                for (int j = 0; j < 8; ++j) { pa[j] = x[j]; pa[8 + j] = y[j]; }
            }
            // ---- P @ V over these 32 keys, 4 d-tiles of 16 ----
#pragma unroll
            for (int dt = 0; dt < 4; ++dt) {
                const int n = dt * 16 + ln;            // output d column
                v16bf vb;
                {
                    const __bf16* p0 = &vT[n * 128 + kt + lg * 8];
                    v8bf a = *(const v8bf*)(p0);
                    v8bf b = *(const v8bf*)(p0 + 16);
#pragma unroll
                    for (int j = 0; j < 8; ++j) { vb[j] = a[j]; vb[8 + j] = b[j]; }
                }
                acc[dt] = wmma_bf16(pa, vb, acc[dt]);
            }
        }
    }

    // finalize + store [16 q x 64 d] strip as f32
    float inv[8];
#pragma unroll
    for (int r = 0; r < 8; ++r) inv[r] = 1.f / lrow[r];
#pragma unroll
    for (int dt = 0; dt < 4; ++dt)
#pragma unroll
        for (int r = 0; r < 8; ++r) {
            const int row = m * CC + q0 + r + 8 * lg;
            const int col = h * DHH + dt * 16 + ln;
            out[(size_t)row * HH + col] = acc[dt][r] * inv[r];
        }
}

// ---------------------------------------------------------------------------
// Softmax over rows of length 256 (one wave per row), in place.
// ---------------------------------------------------------------------------
__global__ __launch_bounds__(256)
void softmax256_kernel(float* __restrict__ x, int nrows) {
    const int lane = threadIdx.x & 31;
    const int wave = threadIdx.x >> 5;
    const int row  = blockIdx.x * 8 + wave;
    if (row >= nrows) return;
    float* p = x + (size_t)row * 256;
    float v[8];
#pragma unroll
    for (int j = 0; j < 8; ++j) v[j] = p[lane + j * 32];
    float mx = v[0];
#pragma unroll
    for (int j = 1; j < 8; ++j) mx = fmaxf(mx, v[j]);
    mx = fmaxf(mx, __shfl_xor(mx, 1));
    mx = fmaxf(mx, __shfl_xor(mx, 2));
    mx = fmaxf(mx, __shfl_xor(mx, 4));
    mx = fmaxf(mx, __shfl_xor(mx, 8));
    mx = fmaxf(mx, __shfl_xor(mx, 16));
    float s = 0.f;
#pragma unroll
    for (int j = 0; j < 8; ++j) { v[j] = __expf(v[j] - mx); s += v[j]; }
    s += __shfl_xor(s, 1);
    s += __shfl_xor(s, 2);
    s += __shfl_xor(s, 4);
    s += __shfl_xor(s, 8);
    s += __shfl_xor(s, 16);
    const float inv = 1.f / s;
#pragma unroll
    for (int j = 0; j < 8; ++j) p[lane + j * 32] = v[j] * inv;
}

// ---------------------------------------------------------------------------
// LayerNorm over rows of 512 (one wave per row), optional residual add.
// ---------------------------------------------------------------------------
__global__ __launch_bounds__(256)
void ln512_kernel(const float* __restrict__ a, const float* __restrict__ res,
                  const float* __restrict__ g, const float* __restrict__ b,
                  float* __restrict__ out, int nrows) {
    const int lane = threadIdx.x & 31;
    const int wave = threadIdx.x >> 5;
    const int row  = blockIdx.x * 8 + wave;
    if (row >= nrows) return;
    const float* pa = a + (size_t)row * 512;
    const float* pr = res ? res + (size_t)row * 512 : nullptr;
    float x[16];
#pragma unroll
    for (int j = 0; j < 16; ++j) {
        const int idx = lane + j * 32;
        x[j] = pa[idx] + (pr ? pr[idx] : 0.f);
    }
    float s = 0.f;
#pragma unroll
    for (int j = 0; j < 16; ++j) s += x[j];
    s += __shfl_xor(s, 1); s += __shfl_xor(s, 2); s += __shfl_xor(s, 4);
    s += __shfl_xor(s, 8); s += __shfl_xor(s, 16);
    const float mu = s * (1.f / 512.f);
    float vs = 0.f;
#pragma unroll
    for (int j = 0; j < 16; ++j) { const float d = x[j] - mu; vs += d * d; }
    vs += __shfl_xor(vs, 1); vs += __shfl_xor(vs, 2); vs += __shfl_xor(vs, 4);
    vs += __shfl_xor(vs, 8); vs += __shfl_xor(vs, 16);
    const float rstd = rsqrtf(vs * (1.f / 512.f) + EPS_LN);
    float* po = out + (size_t)row * 512;
#pragma unroll
    for (int j = 0; j < 16; ++j) {
        const int idx = lane + j * 32;
        po[idx] = (x[j] - mu) * rstd * g[idx] + b[idx];
    }
}

// ---------------------------------------------------------------------------
// Host orchestration. Workspace peak ~211 MiB (linear layout with reuse).
// ---------------------------------------------------------------------------
extern "C" void kernel_launch(void* const* d_in, const int* in_sizes, int n_in,
                              void* d_out, int out_size, void* d_ws, size_t ws_size,
                              hipStream_t stream) {
    const float* q_embed = (const float*)d_in[0];
    const float* ximg    = (const float*)d_in[1];
    const float* gt      = (const float*)d_in[2];
    const float* pred    = (const float*)d_in[3];
    const float* cp_w1   = (const float*)d_in[4];
    const float* cp_b1   = (const float*)d_in[5];
    const float* cp_w2   = (const float*)d_in[6];
    const float* cp_b2   = (const float*)d_in[7];
    const float* in_w    = (const float*)d_in[8];
    const float* in_b    = (const float*)d_in[9];
    const float* out_w   = (const float*)d_in[10];
    const float* out_b   = (const float*)d_in[11];
    const float* lnc_g   = (const float*)d_in[12];
    const float* lnc_b   = (const float*)d_in[13];
    const float* wq_w    = (const float*)d_in[14];
    const float* wq_b    = (const float*)d_in[15];
    const float* wk_w    = (const float*)d_in[16];
    const float* wk_b    = (const float*)d_in[17];
    const float* wv_w    = (const float*)d_in[18];
    const float* wv_b    = (const float*)d_in[19];
    const float* lno_g   = (const float*)d_in[20];
    const float* lno_b   = (const float*)d_in[21];

    char* ws = (char*)d_ws;
    float*  w1p     = (float*)(ws + 0);            // 512*544*4      = 1,114,112 B
    float*  ctxin   = (float*)(ws + 1114112);      // 16384*544*4    = 35,651,584 B
    float*  h1      = (float*)(ws + 36765696);     // 16384*512*4
    float*  ctxemb  = (float*)(ws + 70320128);     // 16384*512*4
    float*  sa_attn = (float*)(ws + 103874560);    // 16384*512*4
    float*  sa_proj = (float*)(ws + 137428992);    // 16384*512*4
    __bf16* qkv     = (__bf16*)(ws + 170983424);   // 16384*1536*2   -> end 221,315,072 B
    // reuse (producers of these regions are dead by the time they're written)
    float* ctx2   = ctxin;                          // over dead ctxin
    float* Qc     = w1p;                            // over dead w1p
    float* K2     = h1;                             // over dead h1
    float* V2     = ctxemb;                         // over dead ctxemb
    float* V2T    = sa_proj;                        // over dead sa_proj
    float* logits = (float*)(ws + 170983424);       // over dead qkv region
    float* z      = sa_attn;                        // over dead sa_attn

    const dim3 blk(256);

    // 0) pack padded inputs
    pack_w1_kernel<<<(HH * KP + 255) / 256, blk, 0, stream>>>(cp_w1, w1p);
    pack_ctxin_kernel<<<(MC_ * KP + 255) / 256, blk, 0, stream>>>(ximg, gt, pred, ctxin);

    // 1) h1 = relu(ctxin @ w1p^T + cp_b1)          [MC,512], K=544
    gemm_wmma_kernel<float><<<dim3(512 / 128, MC_ / 64, 1), blk, 0, stream>>>(
        ctxin, KP, 0, w1p, KP, 0, cp_b1, 1.f, 1, KP, h1, 512, 0);

    // 2) ctx_emb = h1 @ cp_w2^T + cp_b2            [MC,512], K=512
    gemm_wmma_kernel<float><<<dim3(4, MC_ / 64, 1), blk, 0, stream>>>(
        h1, 512, 0, cp_w2, 512, 0, cp_b2, 1.f, 0, 512, ctxemb, 512, 0);

    // 3) qkv(bf16) = ctx_emb @ in_w^T + in_b       [MC,1536]
    gemm_wmma_kernel<__bf16><<<dim3(1536 / 128, MC_ / 64, 1), blk, 0, stream>>>(
        ctxemb, 512, 0, in_w, 512, 0, in_b, 1.f, 0, 512, qkv, 1536, 0);

    // 4) fused self-attention -> sa_attn (f32)     one block per (m, head)
    flash_self_attn_kernel<<<dim3(MM * NHH), dim3(512), 0, stream>>>(qkv, sa_attn);

    // 5) sa_proj = sa_attn @ out_w^T + out_b  (sa_proj aliases V2T later; OK:
    //    written now, consumed by LN1 before V2T is produced)
    gemm_wmma_kernel<float><<<dim3(4, MC_ / 64, 1), blk, 0, stream>>>(
        sa_attn, 512, 0, out_w, 512, 0, out_b, 1.f, 0, 512, sa_proj, 512, 0);

    // 6) ctx2 = LN(ctx_emb + sa_proj)
    ln512_kernel<<<MC_ / 8, blk, 0, stream>>>(ctxemb, sa_proj, lnc_g, lnc_b, ctx2, MC_);

    // 7) Q = query_embed @ wq^T + wq_b             [256,512]
    gemm_wmma_kernel<float><<<dim3(4, BB / 64, 1), blk, 0, stream>>>(
        q_embed, 512, 0, wq_w, 512, 0, wq_b, 1.f, 0, 512, Qc, 512, 0);

    // 8) K2 = ctx_img_feat @ wk^T + wk_b           [MC,512]
    gemm_wmma_kernel<float><<<dim3(4, MC_ / 64, 1), blk, 0, stream>>>(
        ximg, 512, 0, wk_w, 512, 0, wk_b, 1.f, 0, 512, K2, 512, 0);

    // 9) V2 = ctx2 @ wv^T + wv_b                   [MC,512]
    gemm_wmma_kernel<float><<<dim3(4, MC_ / 64, 1), blk, 0, stream>>>(
        ctx2, 512, 0, wv_w, 512, 0, wv_b, 1.f, 0, 512, V2, 512, 0);

    // 9b) V2T[m][h][c] = V2[m][c][h]  (so attn @ V GEMM has contiguous B)
    transpose_256x512_kernel<<<dim3(HH / 32, CC / 32, MM), dim3(32, 8), 0, stream>>>(
        V2, V2T);

    // 10) logits = (Q @ K2^T) * H^-0.5             [256, 16384]
    gemm_wmma_kernel<float><<<dim3(MC_ / 128, BB / 64, 1), blk, 0, stream>>>(
        Qc, 512, 0, K2, 512, 0, nullptr, 0.04419417382415922f, 0, 512,
        logits, MC_, 0);

    // 11) softmax over c (rows of 256), in place
    softmax256_kernel<<<(BB * MM) / 8, blk, 0, stream>>>(logits, BB * MM);

    // 12) z[b,m,:] = attn[b,m,:] @ V2[m]           batched over m (grid.z)
    //     A: logits, lda=16384, batch stride C; W = V2T[m] ([512 h][256 c])
    gemm_wmma_kernel<float><<<dim3(512 / 128, BB / 64, MM), blk, 0, stream>>>(
        logits, MC_, CC, V2T, CC, HH * CC, nullptr, 1.f, 0, CC,
        z, MM * HH, HH);

    // 13) out = LN(z)                              [B*M, 512] -> d_out
    ln512_kernel<<<(BB * MM) / 8, blk, 0, stream>>>(z, nullptr, lno_g, lno_b,
                                                    (float*)d_out, BB * MM);
}
